// Encoder_66322884984933
// MI455X (gfx1250) — matrix operand order repaired and checked by reading, not compiled, not run on hardware
//
#include <hip/hip_runtime.h>
#include <hip/hip_bf16.h>
#include <math.h>

// ---------------------------------------------------------------------------
// GCN encoder for MI455X (gfx1250): bf16 WMMA (16x16x32) with f32 accumulate.
// ---------------------------------------------------------------------------

typedef __attribute__((ext_vector_type(16))) __bf16 v16bf;
typedef __attribute__((ext_vector_type(8)))  float  v8f;

union Frag16 {
    unsigned u[8];
    v16bf    v;
};

__device__ __forceinline__ unsigned short f32_to_bf16(float x) {
    union { float f; unsigned u; } c; c.f = x;
    unsigned r = c.u + 0x7FFFu + ((c.u >> 16) & 1u);   // round-to-nearest-even
    return (unsigned short)(r >> 16);
}

#define Bn 256
#define Nn 256
#define Fn 128

// ---------------------------------------------------------------------------
// Kernel 1: degree / D^{-1/2}.  One wave32 per adjacency row.
// rowsum = sum_m adj[n,m] + 1 - adj[n,n];  d = rsqrt(max(rowsum,1))
// ---------------------------------------------------------------------------
__global__ void __launch_bounds__(256) k_degree(const float* __restrict__ adj,
                                                float* __restrict__ deg) {
    int row  = blockIdx.x * 8 + (threadIdx.x >> 5);   // [0, B*N)
    int lane = threadIdx.x & 31;
    const float* arow = adj + (size_t)row * Nn;
    float s = 0.f;
#pragma unroll
    for (int k = 0; k < 8; ++k) s += arow[lane + 32 * k];
#pragma unroll
    for (int off = 16; off; off >>= 1) s += __shfl_xor(s, off, 32);
    if (lane == 0) {
        int n = row & (Nn - 1);
        s += 1.0f - arow[n];             // replace diagonal with 1
        s = fmaxf(s, 1.0f);
        deg[row] = rsqrtf(s);
    }
}

// ---------------------------------------------------------------------------
// Kernel 2: materialize normalized adjacency in bf16.
// As[b,n,m] = d[b,n]*d[b,m]*(n==m ? 1 : adj[b,n,m])
// ---------------------------------------------------------------------------
__global__ void __launch_bounds__(256) k_build_adj(const float* __restrict__ adj,
                                                   const float* __restrict__ deg,
                                                   unsigned short* __restrict__ As) {
    int row = blockIdx.x;                 // b*N + n
    int b   = row >> 8;
    int n   = row & (Nn - 1);
    int m   = threadIdx.x;
    float a = adj[(size_t)row * Nn + m];
    if (m == n) a = 1.0f;
    float v = deg[row] * deg[(b << 8) + m] * a;
    As[(size_t)row * Nn + m] = f32_to_bf16(v);
}

// ---------------------------------------------------------------------------
// WMMA fragment gathers from LDS (bf16, padded row stride in halfwords).
// A 16x32 bf16: lane half selects kb=0/16; dwords K=kb+2v
// B 32x16 bf16 (stored transposed [n][k]): lane half selects kb2=0/16; dwords K=kb2+2v
// ---------------------------------------------------------------------------
__device__ __forceinline__ void gather_a(Frag16& af, const unsigned* Aru, int ks, int hi) {
    int kh = (ks * 32 + hi * 16) >> 1;         // dword index
#pragma unroll
    for (int v = 0; v < 4; ++v) {
        af.u[v]     = Aru[kh + v];
        af.u[4 + v] = Aru[kh + 4 + v];
    }
}
__device__ __forceinline__ void gather_b(Frag16& bf, const unsigned* Bru, int ks, int hi) {
    int kh2 = (ks * 32 + hi * 16) >> 1;
#pragma unroll
    for (int v = 0; v < 8; ++v) bf.u[v] = Bru[kh2 + v];
}

// ---------------------------------------------------------------------------
// Kernel 3: U = A(f32, Mx128) @ W(f32, 128x128) -> bf16.  M = B*N = 65536.
// Block = 256 threads (8 waves); block tile 128x128; wave tile 16x128.
// ---------------------------------------------------------------------------
#define LDAW 136   // 128 + 8 halfword pad (bank-conflict spread)
__global__ void __launch_bounds__(256)
k_gemm_weight(const float* __restrict__ A, const float* __restrict__ W,
              unsigned short* __restrict__ Uout) {
    __shared__ __align__(16) unsigned short ldsA[128 * LDAW];
    __shared__ __align__(16) unsigned short ldsWt[128 * LDAW];   // transposed [n][k]
    const int t    = threadIdx.x;
    const int lane = t & 31;
    const int w    = t >> 5;
    const size_t rowBase = (size_t)blockIdx.x * 128;

    // stage W transposed (full K=128), converting f32 -> bf16
#pragma unroll 4
    for (int i = 0; i < 64; ++i) {
        int idx = t + 256 * i;
        int k = idx >> 7, n = idx & 127;
        ldsWt[n * LDAW + k] = f32_to_bf16(W[k * Fn + n]);
    }
    // stage A tile 128x128, f32 -> bf16
#pragma unroll 4
    for (int i = 0; i < 64; ++i) {
        int idx = t + 256 * i;
        int r = idx >> 7, c = idx & 127;
        ldsA[r * LDAW + c] = f32_to_bf16(A[(rowBase + r) * Fn + c]);
    }
    __syncthreads();

    v8f acc[8];
#pragma unroll
    for (int ct = 0; ct < 8; ++ct)
#pragma unroll
        for (int j = 0; j < 8; ++j) acc[ct][j] = 0.0f;

    const int m  = lane & 15;
    const int hi = (lane & 16) ? 1 : 0;
    const unsigned* Aru = (const unsigned*)&ldsA[(w * 16 + m) * LDAW];

#pragma unroll
    for (int ks = 0; ks < 4; ++ks) {              // K = 4 * 32 = 128
        Frag16 af;
        gather_a(af, Aru, ks, hi);
#pragma unroll
        for (int ct = 0; ct < 8; ++ct) {
            Frag16 bf;
            const unsigned* Bru = (const unsigned*)&ldsWt[(ct * 16 + m) * LDAW];
            gather_b(bf, Bru, ks, hi);
            acc[ct] = __builtin_amdgcn_wmma_f32_16x16x32_bf16(
                false, af.v, false, bf.v, (short)0, acc[ct], false, false);
        }
    }

    // D layout: dword v -> M = v + hi*8, N = lane&15
#pragma unroll
    for (int ct = 0; ct < 8; ++ct) {
#pragma unroll
        for (int v = 0; v < 8; ++v) {
            size_t r = rowBase + w * 16 + v + hi * 8;
            int    c = ct * 16 + m;
            Uout[r * Fn + c] = f32_to_bf16(acc[ct][v]);
        }
    }
}

// ---------------------------------------------------------------------------
// Kernel 4: batched H[b] = As[b](bf16, 256x256) @ U[b](bf16, 256x128) + bias -> f32
// Grid = B * 2 (row halves).  Block 256 threads; K tiled in chunks of 64.
// ---------------------------------------------------------------------------
#define LDAK 72    // 64 + 8 halfword pad
__global__ void __launch_bounds__(256)
k_gemm_adj(const unsigned short* __restrict__ As, const unsigned short* __restrict__ U,
           const float* __restrict__ bias, float* __restrict__ H) {
    __shared__ __align__(16) unsigned short ldsA[128 * LDAK];
    __shared__ __align__(16) unsigned short ldsUt[128 * LDAK];   // transposed [f][m]
    const int t    = threadIdx.x;
    const int lane = t & 31;
    const int w    = t >> 5;
    const int b    = blockIdx.x >> 1;
    const int rh   = blockIdx.x & 1;

    const unsigned short* Asb = As + (size_t)b * Nn * Nn + (size_t)rh * 128 * Nn;
    const unsigned short* Ub  = U  + (size_t)b * Nn * Fn;
    float*                Hb  = H  + (size_t)b * Nn * Fn + (size_t)rh * 128 * Fn;

    const int m  = lane & 15;
    const int hi = (lane & 16) ? 1 : 0;

    v8f acc[8];
#pragma unroll
    for (int ct = 0; ct < 8; ++ct)
#pragma unroll
        for (int j = 0; j < 8; ++j) acc[ct][j] = 0.0f;

    for (int kc = 0; kc < 4; ++kc) {              // K chunks of 64 (total 256)
        // stage As rows (128 x 64 bf16) as dwords
        const unsigned* Asu = (const unsigned*)Asb;   // row stride 128 dwords
        unsigned* ldsAu = (unsigned*)ldsA;
#pragma unroll 4
        for (int i = 0; i < 16; ++i) {
            int idx = t + 256 * i;                // 0..4095
            int r = idx >> 5, c = idx & 31;
            ldsAu[r * (LDAK / 2) + c] = Asu[r * 128 + kc * 32 + c];
        }
        // stage U chunk transposed: ldsUt[f][k] = U[kc*64+k][f]
        const unsigned* Uu = (const unsigned*)Ub;     // row stride 64 dwords
#pragma unroll 4
        for (int i = 0; i < 16; ++i) {
            int idx = t + 256 * i;                // 0..4095
            int k  = idx >> 6, fu = idx & 63;
            unsigned pv = Uu[(size_t)(kc * 64 + k) * 64 + fu];
            ldsUt[(2 * fu)     * LDAK + k] = (unsigned short)(pv & 0xffffu);
            ldsUt[(2 * fu + 1) * LDAK + k] = (unsigned short)(pv >> 16);
        }
        __syncthreads();

        const unsigned* Aru = (const unsigned*)&ldsA[(w * 16 + m) * LDAK];
#pragma unroll
        for (int ks = 0; ks < 2; ++ks) {          // 2 * 32 = 64 per chunk
            Frag16 af;
            gather_a(af, Aru, ks, hi);
#pragma unroll
            for (int ct = 0; ct < 8; ++ct) {
                Frag16 bf;
                const unsigned* Bru = (const unsigned*)&ldsUt[(ct * 16 + m) * LDAK];
                gather_b(bf, Bru, ks, hi);
                acc[ct] = __builtin_amdgcn_wmma_f32_16x16x32_bf16(
                    false, af.v, false, bf.v, (short)0, acc[ct], false, false);
            }
        }
        __syncthreads();
    }

#pragma unroll
    for (int ct = 0; ct < 8; ++ct) {
        int c = ct * 16 + m;
        float bc = bias[c];
#pragma unroll
        for (int v = 0; v < 8; ++v) {
            int r = w * 16 + v + hi * 8;
            Hb[(size_t)r * Fn + c] = acc[ct][v] + bc;
        }
    }
}

// ---------------------------------------------------------------------------
// Kernel 5: mean/max pool over N, then out = [mean,max] @ Wr + br.
// One block (128 threads) per batch element.
// ---------------------------------------------------------------------------
__global__ void __launch_bounds__(128)
k_pool(const float* __restrict__ H, const float* __restrict__ Wr,
       const float* __restrict__ br, float* __restrict__ out) {
    __shared__ float pooled[2 * Fn];
    int b = blockIdx.x;
    int f = threadIdx.x;                          // 0..127
    const float* Hb = H + (size_t)b * Nn * Fn;
    float s = 0.f, mx = -INFINITY;
    for (int n = 0; n < Nn; ++n) {
        float v = Hb[(size_t)n * Fn + f];
        s += v;
        mx = fmaxf(mx, v);
    }
    pooled[f]      = s * (1.0f / (float)Nn);
    pooled[Fn + f] = mx;
    __syncthreads();
    float acc = br[f];
    for (int k = 0; k < 2 * Fn; ++k) acc += pooled[k] * Wr[k * Fn + f];
    out[(size_t)b * Fn + f] = acc;
}

// ---------------------------------------------------------------------------
// Launcher.  Inputs: x, adj, W1, b1, W2, b2, Wr, br  (all f32).
// Workspace layout: deg f32 | As bf16 | U bf16 | H f32   (~84 MB total)
// ---------------------------------------------------------------------------
extern "C" void kernel_launch(void* const* d_in, const int* in_sizes, int n_in,
                              void* d_out, int out_size, void* d_ws, size_t ws_size,
                              hipStream_t stream) {
    (void)in_sizes; (void)n_in; (void)out_size; (void)ws_size;
    const float* x   = (const float*)d_in[0];
    const float* adj = (const float*)d_in[1];
    const float* W1  = (const float*)d_in[2];
    const float* b1  = (const float*)d_in[3];
    const float* W2  = (const float*)d_in[4];
    const float* b2  = (const float*)d_in[5];
    const float* Wr  = (const float*)d_in[6];
    const float* br  = (const float*)d_in[7];
    float* out = (float*)d_out;

    char* ws = (char*)d_ws;
    const size_t deg_bytes = (size_t)Bn * Nn * 4;                 // 256 KB
    const size_t As_bytes  = (size_t)Bn * Nn * Nn * 2;            // 33.5 MB
    const size_t U_bytes   = (size_t)Bn * Nn * Fn * 2;            // 16.8 MB
    float*          deg = (float*)ws;
    unsigned short* As  = (unsigned short*)(ws + deg_bytes);
    unsigned short* U   = (unsigned short*)(ws + deg_bytes + As_bytes);
    float*          H   = (float*)(ws + deg_bytes + As_bytes + U_bytes);

    k_degree   <<<dim3(Bn * Nn / 8), dim3(256), 0, stream>>>(adj, deg);
    k_build_adj<<<dim3(Bn * Nn),     dim3(256), 0, stream>>>(adj, deg, As);

    // Layer 1: U = X @ W1 ; H = As @ U + b1
    k_gemm_weight<<<dim3(Bn * Nn / 128), dim3(256), 0, stream>>>(x, W1, U);
    k_gemm_adj   <<<dim3(Bn * 2),        dim3(256), 0, stream>>>(As, U, b1, H);

    // Layer 2: U = H @ W2 ; H = As @ U + b2
    k_gemm_weight<<<dim3(Bn * Nn / 128), dim3(256), 0, stream>>>(H, W2, U);
    k_gemm_adj   <<<dim3(Bn * 2),        dim3(256), 0, stream>>>(As, U, b2, H);

    // Pool + readout
    k_pool<<<dim3(Bn), dim3(128), 0, stream>>>(H, Wr, br, out);
}
